// MultiHeadAttention_72791105733046
// MI455X (gfx1250) — matrix operand order repaired and checked by reading, not compiled, not run on hardware
//
#include <hip/hip_runtime.h>
#include <hip/hip_bf16.h>

// ---------------------------------------------------------------------------
// Problem constants (from reference): B=4, S=2048, D=1024, H=16, DH=64
// ---------------------------------------------------------------------------
#define BB 4
#define SS 2048
#define DD 1024
#define HH 16
#define DHH 64
#define MM (BB * SS)        // 8192 rows
#define THETA_F 10000.0f

typedef __bf16 bf16_t;
typedef __attribute__((ext_vector_type(8)))  __bf16 v8bf;
typedef __attribute__((ext_vector_type(16))) __bf16 v16bf;
typedef __attribute__((ext_vector_type(8)))  float  v8f;

union BF16x16 { v16bf v; v8bf h[2]; };

static __device__ __forceinline__ v8f wmma_bf16(v16bf a, v16bf b, v8f c) {
  return __builtin_amdgcn_wmma_f32_16x16x32_bf16(
      /*neg_a=*/false, a, /*neg_b=*/false, b,
      /*c_mod=*/(short)0, c, /*reuse_a=*/false, /*reuse_b=*/false);
}

// ---------------------------------------------------------------------------
// 1) LayerNorm: x (f32, row of 1024) -> xn (bf16 row-major)
// ---------------------------------------------------------------------------
__global__ void ln_kernel(const float* __restrict__ x,
                          const float* __restrict__ gamma,
                          const float* __restrict__ beta,
                          __hip_bfloat16* __restrict__ xn) {
  const int row = blockIdx.x;
  const float* xr = x + (size_t)row * DD;
  float s = 0.f, ss = 0.f;
  for (int i = threadIdx.x; i < DD; i += 256) {
    float v = xr[i]; s += v; ss += v * v;
  }
  for (int off = 16; off >= 1; off >>= 1) {
    s  += __shfl_xor(s,  off);
    ss += __shfl_xor(ss, off);
  }
  __shared__ float sh[16];
  const int w = threadIdx.x >> 5;
  if ((threadIdx.x & 31) == 0) { sh[w] = s; sh[w + 8] = ss; }
  __syncthreads();
  if (threadIdx.x == 0) {
    float S = 0.f, SSu = 0.f;
    for (int i = 0; i < 8; ++i) { S += sh[i]; SSu += sh[i + 8]; }
    sh[0] = S; sh[8] = SSu;
  }
  __syncthreads();
  const float mu  = sh[0] * (1.f / DD);
  const float var = sh[8] * (1.f / DD) - mu * mu;
  const float inv = rsqrtf(var + 1e-5f);
  for (int i = threadIdx.x; i < DD; i += 256) {
    float v = (xr[i] - mu) * inv * gamma[i] + beta[i];
    xn[(size_t)row * DD + i] = __float2bfloat16(v);
  }
}

// ---------------------------------------------------------------------------
// 2) Pack a 1024x1024 f32 weight (K x N) into WMMA B-fragment order (bf16).
// Fragment (kt,nt): 32 lanes x 16 bf16. h=lane/16: K rows kt*32+8h+j (j<8),
// kt*32+16+8h+(j-8) (j>=8); N=nt*16+(lane&15).  (ISA 16-bit B layout)
// ---------------------------------------------------------------------------
__global__ void pack_w_kernel(const float* __restrict__ w,
                              __hip_bfloat16* __restrict__ dst) {
  const int idx  = blockIdx.x * 256 + threadIdx.x;   // 1024*1024 total
  const int j    = idx & 15;
  const int lane = (idx >> 4) & 31;
  const int f    = idx >> 9;        // fragment id = nt*32 + kt  (KT = 32)
  const int kt   = f & 31;
  const int nt   = f >> 5;
  const int K = kt * 32 + ((lane >> 4) << 3) + (j & 7) + ((j >> 3) << 4);
  const int N = nt * 16 + (lane & 15);
  dst[idx] = __float2bfloat16(w[(size_t)K * DD + N]);
}

// ---------------------------------------------------------------------------
// 3) QKV GEMM: xn(8192x1024) @ [Wq|Wk|Wv]packed -> q,k in (B*H,S,DH) bf16,
//    v transposed into (B*H,DH,S) bf16.  One wave = 32(M) x 64(N) tile.
// ---------------------------------------------------------------------------
__global__ void gemm_qkv_kernel(const __hip_bfloat16* __restrict__ xn,
                                const __hip_bfloat16* __restrict__ wpack, // 3 regions of 1M
                                const float* __restrict__ bq,
                                const float* __restrict__ bk,
                                const float* __restrict__ bv,
                                __hip_bfloat16* __restrict__ qb,
                                __hip_bfloat16* __restrict__ kb,
                                __hip_bfloat16* __restrict__ vtb) {
  const int lane    = threadIdx.x & 31;
  const int wave    = threadIdx.x >> 5;
  const int laneRow = lane & 15;
  const int hi      = (lane & 16) ? 1 : 0;
  const int mBase   = blockIdx.x * 128 + wave * 32;
  const int nBase   = blockIdx.y * 64;              // 0..3071
  const int sel     = nBase / DD;                   // 0=q 1=k 2=v (uniform per block)
  const int nLoc    = nBase % DD;

  const __hip_bfloat16* arow0 = xn + (size_t)(mBase + laneRow) * DD;
  const __hip_bfloat16* arow1 = arow0 + (size_t)16 * DD;
  const __hip_bfloat16* bbase = wpack + (size_t)sel * (DD * DD);
  const int aoff = hi ? 8 : 0;

  v8f acc[2][4] = {};
  for (int kt = 0; kt < 32; ++kt) {
    const int k0 = kt * 32;
    BF16x16 a0, a1;
    a0.h[0] = *(const v8bf*)(arow0 + k0 + aoff);
    a0.h[1] = *(const v8bf*)(arow0 + k0 + aoff + 16);
    a1.h[0] = *(const v8bf*)(arow1 + k0 + aoff);
    a1.h[1] = *(const v8bf*)(arow1 + k0 + aoff + 16);
#pragma unroll
    for (int nt = 0; nt < 4; ++nt) {
      const int ntG = (nLoc >> 4) + nt;
      const __hip_bfloat16* bp = bbase + (((size_t)(ntG * 32 + kt)) * 32 + lane) * 16;
      BF16x16 bf;
      bf.h[0] = *(const v8bf*)(bp);
      bf.h[1] = *(const v8bf*)(bp + 8);
      acc[0][nt] = wmma_bf16(a0.v, bf.v, acc[0][nt]);
      acc[1][nt] = wmma_bf16(a1.v, bf.v, acc[1][nt]);
    }
  }

  const float* bias3[3] = { bq, bk, bv };
#pragma unroll
  for (int mt = 0; mt < 2; ++mt) {
#pragma unroll
    for (int nt = 0; nt < 4; ++nt) {
      const int nn   = nLoc + nt * 16 + laneRow;   // 0..1023
      const int h    = nn / DHH;
      const int dh   = nn % DHH;
      const float bi = bias3[sel][nn];
      const int rowTop = mBase + mt * 16 + hi * 8;
#pragma unroll
      for (int r = 0; r < 8; ++r) {
        const int row = rowTop + r;
        const int b   = row >> 11;       // /S
        const int s   = row & (SS - 1);
        const int bh  = b * HH + h;
        const __hip_bfloat16 hv = __float2bfloat16(acc[mt][nt][r] + bi);
        if (sel == 2)      vtb[((size_t)bh * DHH + dh) * SS + s] = hv;
        else if (sel == 0) qb[((size_t)bh * SS + s) * DHH + dh]  = hv;
        else               kb[((size_t)bh * SS + s) * DHH + dh]  = hv;
      }
    }
  }
}

// ---------------------------------------------------------------------------
// 4) Interleaved RoPE on q,k in (B*H,S,DH); one thread per (bh,s,pair)
// ---------------------------------------------------------------------------
__global__ void rope_kernel(__hip_bfloat16* __restrict__ qb,
                            __hip_bfloat16* __restrict__ kb) {
  const int idx = blockIdx.x * 256 + threadIdx.x;  // B*H*S*(DH/2) = 4,194,304
  const int i   = idx & 31;
  const int s   = (idx >> 5) & (SS - 1);
  const int bh  = idx >> 16;
  const size_t base = ((size_t)bh * SS + s) * DHH + 2 * i;
  const float fr = (float)s * __powf(THETA_F, -(float)(2 * i) / (float)DHH);
  float sn, cs;
  __sincosf(fr, &sn, &cs);
  float q1 = __bfloat162float(qb[base]), q2 = __bfloat162float(qb[base + 1]);
  qb[base]     = __float2bfloat16(q1 * cs - q2 * sn);
  qb[base + 1] = __float2bfloat16(q1 * sn + q2 * cs);
  float k1 = __bfloat162float(kb[base]), k2 = __bfloat162float(kb[base + 1]);
  kb[base]     = __float2bfloat16(k1 * cs - k2 * sn);
  kb[base + 1] = __float2bfloat16(k1 * sn + k2 * cs);
}

// ---------------------------------------------------------------------------
// 5) Flash attention. One wave (32 threads) per (bh, 16-row q tile).
//    Per 32-key step: 4 WMMAs for Q@K^T, online softmax, P->bf16 via LDS
//    transpose (per-wave 1KB), 4 WMMAs for P@V.  Output bf16 (B,S,H*DH).
// ---------------------------------------------------------------------------
__global__ void attn_kernel(const __hip_bfloat16* __restrict__ qb,
                            const __hip_bfloat16* __restrict__ kb,
                            const __hip_bfloat16* __restrict__ vtb,
                            const int* __restrict__ lengths,
                            __hip_bfloat16* __restrict__ attn) {
  const int lane    = threadIdx.x & 31;
  const int laneRow = lane & 15;
  const int hi      = (lane & 16) ? 1 : 0;
  const int qt = blockIdx.x;
  const int bh = blockIdx.y;
  const int b  = bh >> 4;
  const int h  = bh & 15;
  const int len = lengths[b];

  __shared__ __align__(16) __hip_bfloat16 pbuf[16 * 32];

  // Q A-fragments (16x32 each, covering dh 0..31 and 32..63)
  const __hip_bfloat16* qrow = qb + ((size_t)bh * SS + qt * 16 + laneRow) * DHH;
  const int aoff = hi ? 8 : 0;
  BF16x16 aq0, aq1;
  aq0.h[0] = *(const v8bf*)(qrow + aoff);
  aq0.h[1] = *(const v8bf*)(qrow + aoff + 16);
  aq1.h[0] = *(const v8bf*)(qrow + 32 + aoff);
  aq1.h[1] = *(const v8bf*)(qrow + 32 + aoff + 16);

  v8f o[4] = {};
  float mrow[8], lrow[8];
#pragma unroll
  for (int r = 0; r < 8; ++r) { mrow[r] = -__builtin_inff(); lrow[r] = 0.f; }

  const float scale = 0.125f;                 // 1/sqrt(64)
  const int ktMax = (qt * 16 + 15) >> 5;      // causal loop bound
  for (int kt = 0; kt <= ktMax; ++kt) {
    const int kk = kt * 32;
    float sc[2][8];
#pragma unroll
    for (int kg = 0; kg < 2; ++kg) {
      const int keyCol = kk + kg * 16 + laneRow;    // key this lane covers (N dim)
      const __hip_bfloat16* krow = kb + ((size_t)bh * SS + keyCol) * DHH;
      const int boff = hi ? 8 : 0;
      BF16x16 bk0, bk1;
      bk0.h[0] = *(const v8bf*)(krow + boff);
      bk0.h[1] = *(const v8bf*)(krow + boff + 16);
      bk1.h[0] = *(const v8bf*)(krow + 32 + boff);
      bk1.h[1] = *(const v8bf*)(krow + 32 + boff + 16);
      v8f s8 = {};
      s8 = wmma_bf16(aq0.v, bk0.v, s8);
      s8 = wmma_bf16(aq1.v, bk1.v, s8);
#pragma unroll
      for (int r = 0; r < 8; ++r) {
        const int qIdx = qt * 16 + r + hi * 8;
        const bool valid = (keyCol <= qIdx) && (keyCol < len);
        sc[kg][r] = valid ? s8[r] * scale : -__builtin_inff();
      }
    }
    // online softmax update over the 32-key slab
#pragma unroll
    for (int r = 0; r < 8; ++r) {
      float t = fmaxf(sc[0][r], sc[1][r]);
      t = fmaxf(t, __shfl_xor(t, 1));
      t = fmaxf(t, __shfl_xor(t, 2));
      t = fmaxf(t, __shfl_xor(t, 4));
      t = fmaxf(t, __shfl_xor(t, 8));
      const float mn  = fmaxf(mrow[r], t);
      const float fac = __expf(mrow[r] - mn);
      const float p0  = __expf(sc[0][r] - mn);
      const float p1  = __expf(sc[1][r] - mn);
      float rs = p0 + p1;
      rs += __shfl_xor(rs, 1);
      rs += __shfl_xor(rs, 2);
      rs += __shfl_xor(rs, 4);
      rs += __shfl_xor(rs, 8);
      lrow[r] = lrow[r] * fac + rs;
      mrow[r] = mn;
#pragma unroll
      for (int nt = 0; nt < 4; ++nt) o[nt][r] *= fac;
      const int prow = r + hi * 8;
      pbuf[prow * 32 + laneRow]      = __float2bfloat16(p0);
      pbuf[prow * 32 + 16 + laneRow] = __float2bfloat16(p1);
    }
    asm volatile("s_wait_dscnt 0" ::: "memory");   // same-wave LDS store->load
    // P as A-fragment (16 rows x 32 keys), via LDS transpose
    BF16x16 pa;
    {
      const __hip_bfloat16* pr = &pbuf[laneRow * 32 + (hi ? 8 : 0)];
      pa.h[0] = *(const v8bf*)(pr);
      pa.h[1] = *(const v8bf*)(pr + 16);
    }
    // V B-fragments from transposed V: rows of Vt are contiguous in s
#pragma unroll
    for (int nt = 0; nt < 4; ++nt) {
      const int dh = nt * 16 + laneRow;
      const __hip_bfloat16* vp = vtb + ((size_t)bh * DHH + dh) * SS + kk + (hi ? 8 : 0);
      BF16x16 bv_;
      bv_.h[0] = *(const v8bf*)(vp);
      bv_.h[1] = *(const v8bf*)(vp + 16);
      o[nt] = wmma_bf16(pa.v, bv_.v, o[nt]);
    }
  }

  // Normalize and store to (B, S, H*DH) bf16
#pragma unroll
  for (int nt = 0; nt < 4; ++nt) {
    const int col = h * DHH + nt * 16 + laneRow;
#pragma unroll
    for (int r = 0; r < 8; ++r) {
      const int s = qt * 16 + r + hi * 8;
      const float val = o[nt][r] / lrow[r];
      attn[((size_t)(b * SS + s)) * DD + col] = __float2bfloat16(val);
    }
  }
}

// ---------------------------------------------------------------------------
// 6) Output projection: attn(8192x1024)bf16 @ Wo_packed + bo + x -> out f32
// ---------------------------------------------------------------------------
__global__ void gemm_proj_kernel(const __hip_bfloat16* __restrict__ attn,
                                 const __hip_bfloat16* __restrict__ wpackO,
                                 const float* __restrict__ bo,
                                 const float* __restrict__ x,
                                 float* __restrict__ out) {
  const int lane    = threadIdx.x & 31;
  const int wave    = threadIdx.x >> 5;
  const int laneRow = lane & 15;
  const int hi      = (lane & 16) ? 1 : 0;
  const int mBase   = blockIdx.x * 128 + wave * 32;
  const int nBase   = blockIdx.y * 64;

  const __hip_bfloat16* arow0 = attn + (size_t)(mBase + laneRow) * DD;
  const __hip_bfloat16* arow1 = arow0 + (size_t)16 * DD;
  const int aoff = hi ? 8 : 0;

  v8f acc[2][4] = {};
  for (int kt = 0; kt < 32; ++kt) {
    const int k0 = kt * 32;
    BF16x16 a0, a1;
    a0.h[0] = *(const v8bf*)(arow0 + k0 + aoff);
    a0.h[1] = *(const v8bf*)(arow0 + k0 + aoff + 16);
    a1.h[0] = *(const v8bf*)(arow1 + k0 + aoff);
    a1.h[1] = *(const v8bf*)(arow1 + k0 + aoff + 16);
#pragma unroll
    for (int nt = 0; nt < 4; ++nt) {
      const int ntG = (nBase >> 4) + nt;
      const __hip_bfloat16* bp = wpackO + (((size_t)(ntG * 32 + kt)) * 32 + lane) * 16;
      BF16x16 bf;
      bf.h[0] = *(const v8bf*)(bp);
      bf.h[1] = *(const v8bf*)(bp + 8);
      acc[0][nt] = wmma_bf16(a0.v, bf.v, acc[0][nt]);
      acc[1][nt] = wmma_bf16(a1.v, bf.v, acc[1][nt]);
    }
  }

#pragma unroll
  for (int mt = 0; mt < 2; ++mt) {
#pragma unroll
    for (int nt = 0; nt < 4; ++nt) {
      const int n = nBase + nt * 16 + laneRow;
      const float bi = bo[n];
      const int rowTop = mBase + mt * 16 + hi * 8;
#pragma unroll
      for (int r = 0; r < 8; ++r) {
        const int row = rowTop + r;
        const size_t oidx = (size_t)row * DD + n;
        out[oidx] = acc[mt][nt][r] + bi + x[oidx];
      }
    }
  }
}

// ---------------------------------------------------------------------------
// Launch
// ---------------------------------------------------------------------------
extern "C" void kernel_launch(void* const* d_in, const int* in_sizes, int n_in,
                              void* d_out, int out_size, void* d_ws, size_t ws_size,
                              hipStream_t stream) {
  const float* x       = (const float*)d_in[0];
  const int*   lengths = (const int*)d_in[1];
  const float* Wq = (const float*)d_in[2];
  const float* bq = (const float*)d_in[3];
  const float* Wk = (const float*)d_in[4];
  const float* bk = (const float*)d_in[5];
  const float* Wv = (const float*)d_in[6];
  const float* bv = (const float*)d_in[7];
  const float* Wo = (const float*)d_in[8];
  const float* bo = (const float*)d_in[9];
  const float* gamma = (const float*)d_in[10];
  const float* beta  = (const float*)d_in[11];
  float* out = (float*)d_out;

  // Workspace carve-up (bf16 everywhere). attn aliases xn (dead after QKV GEMM).
  char* ws = (char*)d_ws;
  const size_t eMN  = (size_t)MM * DD;          // 8,388,608
  const size_t eW   = (size_t)DD * DD;          // 1,048,576
  __hip_bfloat16* xn    = (__hip_bfloat16*)ws;                       ws += eMN * 2;  // 16MB
  __hip_bfloat16* wpack = (__hip_bfloat16*)ws;                       ws += 4 * eW * 2; // 8MB (q,k,v,o)
  __hip_bfloat16* qb    = (__hip_bfloat16*)ws;                       ws += eMN * 2;  // 16MB
  __hip_bfloat16* kb    = (__hip_bfloat16*)ws;                       ws += eMN * 2;  // 16MB
  __hip_bfloat16* vtb   = (__hip_bfloat16*)ws;                       ws += eMN * 2;  // 16MB
  __hip_bfloat16* attn  = xn;   // alias: total scratch = 72MB

  // 1) LayerNorm -> bf16
  ln_kernel<<<MM, 256, 0, stream>>>(x, gamma, beta, xn);

  // 2) Pack weights into WMMA B-fragment order
  const int packBlocks = (int)(eW / 256);
  pack_w_kernel<<<packBlocks, 256, 0, stream>>>(Wq, wpack + 0 * eW);
  pack_w_kernel<<<packBlocks, 256, 0, stream>>>(Wk, wpack + 1 * eW);
  pack_w_kernel<<<packBlocks, 256, 0, stream>>>(Wv, wpack + 2 * eW);
  pack_w_kernel<<<packBlocks, 256, 0, stream>>>(Wo, wpack + 3 * eW);

  // 3) Fused QKV GEMM (N spans 3*1024)
  gemm_qkv_kernel<<<dim3(MM / 128, (3 * DD) / 64), 128, 0, stream>>>(
      xn, wpack, bq, bk, bv, qb, kb, vtb);

  // 4) RoPE on q,k
  rope_kernel<<<(BB * HH * SS * (DHH / 2)) / 256, 256, 0, stream>>>(qb, kb);

  // 5) Flash attention (1 wave per block)
  attn_kernel<<<dim3(SS / 16, BB * HH), 32, 0, stream>>>(qb, kb, vtb, lengths, attn);

  // 6) Output projection + bias + residual
  gemm_proj_kernel<<<dim3(MM / 128, DD / 64), 128, 0, stream>>>(
      attn, wpack + 3 * eW, bo, x, out);
}
